// SPGATNet_27101243637897
// MI455X (gfx1250) — matrix operand order repaired and checked by reading, not compiled, not run on hardware
//
#include <hip/hip_runtime.h>
#include <hip/hip_bf16.h>
#include <math.h>

// ---------------- WMMA fragment types (gfx1250, wave32) ----------------
typedef __attribute__((ext_vector_type(16))) _Float16 v16h;
typedef __attribute__((ext_vector_type(8)))  float    v8f;

union FragH { v16h v; unsigned int u[8]; };
union AccF  { v8f  v; float f[8]; };

__device__ __forceinline__ v8f wmma_f32_f16(v16h a, v16h b, v8f c) {
    return __builtin_amdgcn_wmma_f32_16x16x32_f16(false, a, false, b, (short)0, c, false, false);
}

// gfx1250 async global->LDS copy (16B per active lane), tracked by ASYNCcnt
__device__ __forceinline__ void async_load_b128(void* lds, const void* gptr) {
    asm volatile("global_load_async_to_lds_b128 %0, %1, off"
                 :: "v"((unsigned)(unsigned long long)lds),
                    "v"((unsigned long long)gptr)
                 : "memory");
}
__device__ __forceinline__ void wait_async0() {
    asm volatile("s_wait_asynccnt 0" ::: "memory");
}

__device__ __forceinline__ float elu1(float v) { return v > 0.f ? v : (expf(v) - 1.f); }

// ordered-uint encoding for float atomicMax (monotone for all finite floats)
__device__ __forceinline__ unsigned ordenc(float f) {
    unsigned u = __float_as_uint(f);
    return (u & 0x80000000u) ? ~u : (u | 0x80000000u);
}
__device__ __forceinline__ float orddec(unsigned u) {
    unsigned b = (u & 0x80000000u) ? (u & 0x7fffffffu) : ~u;
    return __uint_as_float(b);
}

__device__ __forceinline__ void edge_sd(const int* __restrict__ ei, int idx, int e, int& s, int& d) {
    if (idx < e) { s = ei[idx]; d = ei[e + idx]; }
    else         { s = d = idx - e; }            // implicit self-loops
}

// ---------------- Tiny K=3 encoders ----------------
__global__ __launch_bounds__(256) void k_lin3(const float* __restrict__ x,
                                              const float* __restrict__ W,
                                              const float* __restrict__ b,
                                              float* __restrict__ y, int n, int m) {
    int tid = blockIdx.x * 256 + threadIdx.x;
    if (tid >= n * m) return;
    int i = tid / m, j = tid - i * m;
    const float* xp = x + (size_t)i * 3;
    y[tid] = b[j] + xp[0] * W[j] + xp[1] * W[m + j] + xp[2] * W[2 * m + j];
}
// f16-output variant (GEMM activations)
__global__ __launch_bounds__(256) void k_lin3_f16(const float* __restrict__ x,
                                                  const float* __restrict__ W,
                                                  const float* __restrict__ b,
                                                  _Float16* __restrict__ y, int n, int m) {
    int tid = blockIdx.x * 256 + threadIdx.x;
    if (tid >= n * m) return;
    int i = tid / m, j = tid - i * m;
    const float* xp = x + (size_t)i * 3;
    y[tid] = (_Float16)(b[j] + xp[0] * W[j] + xp[1] * W[m + j] + xp[2] * W[2 * m + j]);
}

// Wt[n, k] = f16(W[k, n]), K padded with zeros to KPad
__global__ __launch_bounds__(256) void k_convert_wt(const float* __restrict__ W,
                                                    _Float16* __restrict__ Wt,
                                                    int K, int N, int KPad) {
    int tid = blockIdx.x * 256 + threadIdx.x;
    if (tid >= N * KPad) return;
    int nn = tid / KPad, k = tid - nn * KPad;
    Wt[tid] = (_Float16)((k < K) ? W[(size_t)k * N + nn] : 0.f);
}

// ---------------- WMMA GEMM: C[M,N] = A[M,K](f16) @ Bt[N,K](f16)^T ----------------
// Block = 128 threads (4 waves), 64x64 tile, K-step 32.
// Tiles staged via async global->LDS B128 copies; stride 40 halves (80B: 16B-aligned,
// 20-bank stride -> conflict-free 16-lane fragment reads).
#define AST 40
__global__ __launch_bounds__(128)
void wmma_gemm_f16(const _Float16* __restrict__ A, const _Float16* __restrict__ Bt,
                   float* __restrict__ C, int M, int N, int K) {
    __shared__ _Float16 As[64][AST];
    __shared__ _Float16 Bs[64][AST];   // Bs[n][k]
    const int t = threadIdx.x;
    const int wave = t >> 5, lane = t & 31;
    const int grp = lane >> 4, l16 = lane & 15;
    const int m0 = blockIdx.x * 64, n0 = blockIdx.y * 64;

    AccF acc[4];
#pragma unroll
    for (int i = 0; i < 4; ++i)
#pragma unroll
        for (int j = 0; j < 8; ++j) acc[i].f[j] = 0.f;

    for (int kk = 0; kk < K; kk += 32) {
        __syncthreads();   // previous iteration's fragment reads done
        // A tile: 64 rows x 32 halfs = 64 rows x 4 x 16B chunks
        for (int idx = t; idx < 64 * 4; idx += 128) {
            int r = idx >> 2, c4 = idx & 3;
            int gr = m0 + r;
            if (gr < M)
                async_load_b128(&As[r][c4 * 8], A + (size_t)gr * K + kk + c4 * 8);
        }
        // B tile: 64 output cols x 32 halfs from transposed weights
        for (int idx = t; idx < 64 * 4; idx += 128) {
            int nl = idx >> 2, c4 = idx & 3;
            async_load_b128(&Bs[nl][c4 * 8], Bt + (size_t)(n0 + nl) * K + kk + c4 * 8);
        }
        wait_async0();
        __syncthreads();

        FragH b;
        {   // B fragment: lane col = l16, K halves 2v at k = grp*16 + 2v
            const _Float16* bp = &Bs[wave * 16 + l16][grp * 16];
#pragma unroll
            for (int v = 0; v < 8; ++v) b.u[v] = *(const unsigned int*)(bp + 2 * v);
        }
#pragma unroll
        for (int mt = 0; mt < 4; ++mt) {
            FragH a;   // A fragment: row = l16, K halves 2v at k = grp*16 + 2v
            const _Float16* ap = &As[mt * 16 + l16][grp * 16];
#pragma unroll
            for (int v = 0; v < 8; ++v) {
                int k0 = 2 * v;
                a.u[v] = *(const unsigned int*)(ap + k0);
            }
            acc[mt].v = wmma_f32_f16(a.v, b.v, acc[mt].v);
        }
    }
    const int gcol = n0 + wave * 16 + l16;
#pragma unroll
    for (int mt = 0; mt < 4; ++mt)
#pragma unroll
        for (int v = 0; v < 8; ++v) {
            int grow = m0 + mt * 16 + grp * 8 + v;
            if (grow < M) C[(size_t)grow * N + gcol] = acc[mt].f[v];
        }
}

// ---------------- GAT attention pieces ----------------
__global__ __launch_bounds__(256) void k_att_scores(const float* __restrict__ g,
                                                    const float* __restrict__ a_s,
                                                    const float* __restrict__ a_d,
                                                    float* __restrict__ asrc,
                                                    float* __restrict__ adst) {
    int i = blockIdx.x;
    int h = threadIdx.x >> 5, lane = threadIdx.x & 31;
    const float* gp = g + (size_t)i * 512 + h * 64;
    float g0 = gp[lane], g1 = gp[lane + 32];
    float s = g0 * a_s[h * 64 + lane] + g1 * a_s[h * 64 + lane + 32];
    float d = g0 * a_d[h * 64 + lane] + g1 * a_d[h * 64 + lane + 32];
#pragma unroll
    for (int m = 16; m > 0; m >>= 1) { s += __shfl_xor(s, m, 32); d += __shfl_xor(d, m, 32); }
    if (lane == 0) { asrc[(size_t)i * 8 + h] = s; adst[(size_t)i * 8 + h] = d; }
}

__global__ __launch_bounds__(256) void k_edge_max(const int* __restrict__ ei,
                                                  const float* __restrict__ asrc,
                                                  const float* __restrict__ adst,
                                                  unsigned* __restrict__ mx, int n, int e) {
    int tid = blockIdx.x * 256 + threadIdx.x;
    if (tid >= (e + n) * 8) return;
    int idx = tid >> 3, h = tid & 7, s, d;
    edge_sd(ei, idx, e, s, d);
    float v = asrc[(size_t)s * 8 + h] + adst[(size_t)d * 8 + h];
    v = v > 0.f ? v : 0.2f * v;                        // leaky_relu(0.2)
    atomicMax(&mx[(size_t)d * 8 + h], ordenc(v));
}

__global__ __launch_bounds__(256) void k_edge_exp(const int* __restrict__ ei,
                                                  const float* __restrict__ asrc,
                                                  const float* __restrict__ adst,
                                                  const unsigned* __restrict__ mx,
                                                  float* __restrict__ sm,
                                                  float* __restrict__ alpha, int n, int e) {
    int tid = blockIdx.x * 256 + threadIdx.x;
    if (tid >= (e + n) * 8) return;
    int idx = tid >> 3, h = tid & 7, s, d;
    edge_sd(ei, idx, e, s, d);
    float v = asrc[(size_t)s * 8 + h] + adst[(size_t)d * 8 + h];
    v = v > 0.f ? v : 0.2f * v;
    float ex = expf(v - orddec(mx[(size_t)d * 8 + h]));
    alpha[(size_t)idx * 8 + h] = ex;
    atomicAdd(&sm[(size_t)d * 8 + h], ex);
}

__global__ __launch_bounds__(256) void k_edge_norm(const int* __restrict__ ei,
                                                   const float* __restrict__ sm,
                                                   float* __restrict__ alpha, int n, int e) {
    int tid = blockIdx.x * 256 + threadIdx.x;
    if (tid >= (e + n) * 8) return;
    int idx = tid >> 3, h = tid & 7, s, d;
    edge_sd(ei, idx, e, s, d);
    alpha[(size_t)idx * 8 + h] /= sm[(size_t)d * 8 + h];
}

// out[d,h,c] += alpha[e,h] * g[s,h,c]  -- 128 threads/edge, float4 per thread
__global__ __launch_bounds__(256) void k_aggregate(const int* __restrict__ ei,
                                                   const float* __restrict__ g,
                                                   const float* __restrict__ alpha,
                                                   float* __restrict__ out, int n, int e) {
    int tid = blockIdx.x * 256 + threadIdx.x;
    if (tid >= (e + n) * 128) return;
    int idx = tid >> 7, q = tid & 127, s, d;
    edge_sd(ei, idx, e, s, d);
    int c = q * 4, h = c >> 6;
    float a = alpha[(size_t)idx * 8 + h];
    const float4 gv = *(const float4*)(g + (size_t)s * 512 + c);
    float* op = out + (size_t)d * 512 + c;
    atomicAdd(op + 0, gv.x * a);
    atomicAdd(op + 1, gv.y * a);
    atomicAdd(op + 2, gv.z * a);
    atomicAdd(op + 3, gv.w * a);
}

// o16 = f16(elu(agg + bc) + res)  (concat layers; result feeds next GEMM only)
__global__ __launch_bounds__(256) void k_finish_concat(const float* __restrict__ agg,
                                                       const float* __restrict__ bc,
                                                       const float* __restrict__ res,
                                                       _Float16* __restrict__ o16, int n) {
    int tid = blockIdx.x * 256 + threadIdx.x;
    if (tid >= n * 512) return;
    float v = agg[tid] + bc[tid & 511];
    o16[tid] = (_Float16)(elu1(v) + res[tid]);
}

// ne = elu(mean_h(agg) + bc3)  (layer 3, mean over heads)
__global__ __launch_bounds__(256) void k_finish_mean(const float* __restrict__ agg,
                                                     const float* __restrict__ bc3,
                                                     float* __restrict__ ne, int n) {
    int tid = blockIdx.x * 256 + threadIdx.x;
    if (tid >= n * 64) return;
    int i = tid >> 6, c = tid & 63;
    float s = 0.f;
#pragma unroll
    for (int h = 0; h < 8; ++h) s += agg[(size_t)i * 512 + h * 64 + c];
    ne[tid] = elu1(s * 0.125f + bc3[c]);
}

// x_out = ne @ Wo + bo  (64 -> 2), warp per node
__global__ __launch_bounds__(256) void k_node_head(const float* __restrict__ ne,
                                                   const float* __restrict__ Wo,
                                                   const float* __restrict__ bo,
                                                   float* __restrict__ out, int n) {
    int node = blockIdx.x * 8 + (threadIdx.x >> 5);
    int lane = threadIdx.x & 31;
    if (node >= n) return;                                   // warp-uniform
    float v0 = ne[(size_t)node * 64 + lane], v1 = ne[(size_t)node * 64 + lane + 32];
    float s0 = v0 * Wo[lane * 2 + 0] + v1 * Wo[(lane + 32) * 2 + 0];
    float s1 = v0 * Wo[lane * 2 + 1] + v1 * Wo[(lane + 32) * 2 + 1];
#pragma unroll
    for (int m = 16; m > 0; m >>= 1) { s0 += __shfl_xor(s0, m, 32); s1 += __shfl_xor(s1, m, 32); }
    if (lane == 0) { out[(size_t)node * 2 + 0] = s0 + bo[0]; out[(size_t)node * 2 + 1] = s1 + bo[1]; }
}

__global__ __launch_bounds__(256) void k_edge_out_init(const float* __restrict__ bm2,
                                                       float* __restrict__ eout, int e) {
    int tid = blockIdx.x * 256 + threadIdx.x;
    if (tid < e * 2) eout[tid] = bm2[tid & 1];
}

// ---------------- Fused edge MLP: relu([ne[src],alpha,ne[dst]] @ Wm1 + bm1) @ Wm2 ----------------
// K=136 padded to 160. Block = 256 threads (8 waves), 16 edge rows, N=256 (wave owns 2 N-tiles).
// Wm1 pre-transposed/padded to f16 [256][160]; tile staged via async B128 copies.
#define KP 160
#define EAST 168
__global__ __launch_bounds__(256)
void k_edge_mlp(const float* __restrict__ ne, const float* __restrict__ alpha,
                const int* __restrict__ ei, const _Float16* __restrict__ Wm1t,
                const float* __restrict__ bm1, const float* __restrict__ Wm2,
                float* __restrict__ eout, int e) {
    __shared__ _Float16 As[16][EAST];
    __shared__ _Float16 Bt[256][AST];
    const int t = threadIdx.x, wave = t >> 5, lane = t & 31;
    const int grp = lane >> 4, l16 = lane & 15;
    const int row0 = blockIdx.x * 16;

    // stage gathered A tile (16 x 160) once (gather + f32->f16: VALU path)
    for (int idx = t; idx < 16 * KP; idx += 256) {
        int r = idx / KP, c = idx - r * KP;
        int row = row0 + r;
        float v = 0.f;
        if (row < e) {
            if (c < 64)       v = ne[(size_t)ei[row] * 64 + c];
            else if (c < 72)  v = alpha[(size_t)row * 8 + (c - 64)];
            else if (c < 136) v = ne[(size_t)ei[e + row] * 64 + (c - 72)];
        }
        As[r][c] = (_Float16)v;
    }

    AccF acc0, acc1;
#pragma unroll
    for (int j = 0; j < 8; ++j) { acc0.f[j] = 0.f; acc1.f[j] = 0.f; }

    for (int kk = 0; kk < KP; kk += 32) {
        __syncthreads();                      // previous Bt consumed (and As stores done)
        // Bt tile: 256 cols x 32 halfs = 256 x 4 x 16B chunks, async
        for (int idx = t; idx < 256 * 4; idx += 256) {
            int nn = idx >> 2, c4 = idx & 3;
            async_load_b128(&Bt[nn][c4 * 8], Wm1t + (size_t)nn * KP + kk + c4 * 8);
        }
        wait_async0();
        __syncthreads();

        FragH a;
        const _Float16* ap = &As[l16][kk + grp * 16];
#pragma unroll
        for (int v = 0; v < 8; ++v) {
            int k0 = 2 * v;
            a.u[v] = *(const unsigned int*)(ap + k0);
        }
#pragma unroll
        for (int s = 0; s < 2; ++s) {
            int ntile = wave * 2 + s;
            FragH b;
            const _Float16* bp = &Bt[ntile * 16 + l16][grp * 16];
#pragma unroll
            for (int v = 0; v < 8; ++v) b.u[v] = *(const unsigned int*)(bp + 2 * v);
            if (s == 0) acc0.v = wmma_f32_f16(a.v, b.v, acc0.v);
            else        acc1.v = wmma_f32_f16(a.v, b.v, acc1.v);
        }
    }

    // epilogue: bias + relu, then x Wm2 (256 -> 2) via shfl reduce + atomics
#pragma unroll
    for (int s = 0; s < 2; ++s) {
        AccF& acc = s ? acc1 : acc0;
        int col = (wave * 2 + s) * 16 + l16;
        float b1 = bm1[col], w20 = Wm2[col * 2 + 0], w21 = Wm2[col * 2 + 1];
        float p0[8], p1[8];
#pragma unroll
        for (int v = 0; v < 8; ++v) {
            float hv = fmaxf(acc.f[v] + b1, 0.f);
            p0[v] = hv * w20; p1[v] = hv * w21;
        }
#pragma unroll
        for (int m = 1; m < 16; m <<= 1)
#pragma unroll
            for (int v = 0; v < 8; ++v) {
                p0[v] += __shfl_xor(p0[v], m, 32);
                p1[v] += __shfl_xor(p1[v], m, 32);
            }
        if (l16 == 0) {
            int rbase = row0 + grp * 8;
#pragma unroll
            for (int v = 0; v < 8; ++v) {
                int rr = rbase + v;
                if (rr < e) {
                    atomicAdd(&eout[(size_t)rr * 2 + 0], p0[v]);
                    atomicAdd(&eout[(size_t)rr * 2 + 1], p1[v]);
                }
            }
        }
    }
}

// ---------------- Host orchestration ----------------
extern "C" void kernel_launch(void* const* d_in, const int* in_sizes, int n_in,
                              void* d_out, int out_size, void* d_ws, size_t ws_size,
                              hipStream_t stream) {
    const float* x   = (const float*)d_in[0];
    const int*   ei  = (const int*)d_in[1];
    const float* Wn  = (const float*)d_in[4];
    const float* bn  = (const float*)d_in[5];
    const float* Wr  = (const float*)d_in[6];
    const float* br  = (const float*)d_in[7];
    const float* W1  = (const float*)d_in[8];
    const float* as1 = (const float*)d_in[9];
    const float* ad1 = (const float*)d_in[10];
    const float* bc1 = (const float*)d_in[11];
    const float* W2  = (const float*)d_in[12];
    const float* as2 = (const float*)d_in[13];
    const float* ad2 = (const float*)d_in[14];
    const float* bc2 = (const float*)d_in[15];
    const float* W3  = (const float*)d_in[16];
    const float* as3 = (const float*)d_in[17];
    const float* ad3 = (const float*)d_in[18];
    const float* bc3 = (const float*)d_in[19];
    const float* Wo  = (const float*)d_in[20];
    const float* bo  = (const float*)d_in[21];
    const float* Wm1 = (const float*)d_in[22];
    const float* bm1 = (const float*)d_in[23];
    const float* Wm2 = (const float*)d_in[24];
    const float* bm2 = (const float*)d_in[25];

    const int n  = in_sizes[0] / 3;   // nodes
    const int e  = in_sizes[2];       // edges
    const int eT = e + n;             // with self-loops

    // workspace layout (float units; f16 buffers 16B-aligned)
    float* ws = (float*)d_ws;
    size_t off = 0;
    float* res   = ws + off; off += (size_t)n * 512;
    float* g     = ws + off; off += (size_t)n * 512;
    float* agg   = ws + off; off += (size_t)n * 512;
    float* asrc  = ws + off; off += (size_t)n * 8;
    float* adst  = ws + off; off += (size_t)n * 8;
    unsigned* mx = (unsigned*)(ws + off); off += (size_t)n * 8;
    float* sm    = ws + off; off += (size_t)n * 8;
    float* alpha = ws + off; off += (size_t)eT * 8;
    float* ne    = ws + off; off += (size_t)n * 64;
    _Float16* h16  = (_Float16*)(ws + off); off += (size_t)n * 32;        // n*64 halves
    _Float16* o16  = (_Float16*)(ws + off); off += (size_t)n * 256;       // n*512 halves
    _Float16* w1t  = (_Float16*)(ws + off); off += (size_t)512 * 32;      // 512x64 halves
    _Float16* w2t  = (_Float16*)(ws + off); off += (size_t)512 * 256;     // 512x512 halves
    _Float16* w3t  = (_Float16*)(ws + off); off += (size_t)512 * 256;
    _Float16* wm1t = (_Float16*)(ws + off); off += (size_t)256 * 80;      // 256x160 halves

    // encoders + weight conversion (f32 -> transposed f16)
    k_lin3_f16<<<(n * 64 + 255) / 256, 256, 0, stream>>>(x, Wn, bn, h16, n, 64);
    k_lin3<<<(n * 512 + 255) / 256, 256, 0, stream>>>(x, Wr, br, res, n, 512);
    k_convert_wt<<<(512 * 64 + 255) / 256, 256, 0, stream>>>(W1, w1t, 64, 512, 64);
    k_convert_wt<<<(512 * 512 + 255) / 256, 256, 0, stream>>>(W2, w2t, 512, 512, 512);
    k_convert_wt<<<(512 * 512 + 255) / 256, 256, 0, stream>>>(W3, w3t, 512, 512, 512);
    k_convert_wt<<<(256 * 160 + 255) / 256, 256, 0, stream>>>(Wm1, wm1t, 136, 256, 160);

    auto run_gat = [&](const _Float16* Ain, int K, const _Float16* Wt,
                       const float* a_s, const float* a_d) {
        dim3 grid((n + 63) / 64, 512 / 64);
        wmma_gemm_f16<<<grid, 128, 0, stream>>>(Ain, Wt, g, n, 512, K);
        k_att_scores<<<n, 256, 0, stream>>>(g, a_s, a_d, asrc, adst);
        hipMemsetAsync(mx, 0, (size_t)n * 8 * 4, stream);
        hipMemsetAsync(sm, 0, (size_t)n * 8 * 4, stream);
        hipMemsetAsync(agg, 0, (size_t)n * 512 * 4, stream);
        int tot8 = eT * 8;
        k_edge_max<<<(tot8 + 255) / 256, 256, 0, stream>>>(ei, asrc, adst, mx, n, e);
        k_edge_exp<<<(tot8 + 255) / 256, 256, 0, stream>>>(ei, asrc, adst, mx, sm, alpha, n, e);
        k_edge_norm<<<(tot8 + 255) / 256, 256, 0, stream>>>(ei, sm, alpha, n, e);
        int totA = eT * 128;
        k_aggregate<<<(totA + 255) / 256, 256, 0, stream>>>(ei, g, alpha, agg, n, e);
    };

    run_gat(h16, 64, w1t, as1, ad1);
    k_finish_concat<<<(n * 512 + 255) / 256, 256, 0, stream>>>(agg, bc1, res, o16, n);
    run_gat(o16, 512, w2t, as2, ad2);
    k_finish_concat<<<(n * 512 + 255) / 256, 256, 0, stream>>>(agg, bc2, res, o16, n);
    run_gat(o16, 512, w3t, as3, ad3);
    k_finish_mean<<<(n * 64 + 255) / 256, 256, 0, stream>>>(agg, bc3, ne, n);

    // node head -> d_out[0 : n*2)
    k_node_head<<<(n + 7) / 8, 256, 0, stream>>>(ne, Wo, bo, (float*)d_out, n);

    // edge MLP -> d_out[n*2 : n*2 + e*2)
    float* eout = (float*)d_out + (size_t)n * 2;
    k_edge_out_init<<<(e * 2 + 255) / 256, 256, 0, stream>>>(bm2, eout, e);
    k_edge_mlp<<<(e + 15) / 16, 256, 0, stream>>>(ne, alpha, ei, wm1t, bm1, Wm2, eout, e);
}
